// FastMHA_67473936220681
// MI455X (gfx1250) — compile-verified, matrix operand order checked by reading, not run on hardware
//
#include <hip/hip_runtime.h>
#include <hip/hip_bf16.h>

// ---------------------------------------------------------------------------
// CDNA5 (gfx1250) bf16-WMMA + TDM implementation of disentangled attention
// B=256, S=128, H=8, DK=128, D_MODEL=1024, P=128
// ---------------------------------------------------------------------------

typedef __attribute__((ext_vector_type(16))) __bf16 v16bf;
typedef __attribute__((ext_vector_type(8)))  float  v8f;
typedef __attribute__((ext_vector_type(4)))  float  v4f;
typedef __attribute__((ext_vector_type(4)))  unsigned int v4u;
typedef __attribute__((ext_vector_type(2)))  unsigned int v2u;
typedef __attribute__((ext_vector_type(4)))  int    v4i;
typedef __attribute__((ext_vector_type(4)))  unsigned int u32x4;
typedef __attribute__((ext_vector_type(8)))  int    i32x8;
typedef __attribute__((ext_vector_type(4)))  int    i32x4;

#define N_B   256
#define N_S   128
#define N_H   8
#define N_DK  128
#define N_DM  1024
#define SCALE_F 0.05103103630798288f   // 1/sqrt(128*3)

// ---------------------------------------------------------------------------
// WMMA helper: D = A(16x32 bf16) x B(32x16 bf16) + C(16x16 f32)
// ---------------------------------------------------------------------------
__device__ __forceinline__ v8f wmma_bf16(v16bf a, v16bf b, v8f c) {
  return __builtin_amdgcn_wmma_f32_16x16x32_bf16(false, a, false, b,
                                                 (short)0, c, false, false);
}

// Load a 16x32 bf16 fragment from a row-major LDS matrix (stride in elements).
// Lane L holds row row0+(L&15); per ISA layout VGPR j holds K pair
// {2j,2j+1}+8*half (j<4) / {16+2(j-4),...}+8*half (j>=4)  ->  two b128 loads.
__device__ __forceinline__ v16bf ldfrag(const __bf16* sm, int stride,
                                        int row0, int col0) {
  int lane = threadIdx.x & 31;
  const v4u* p = (const v4u*)(sm + (row0 + (lane & 15)) * stride + col0 +
                              ((lane >> 4) << 3));
  union { v16bf v; v4u q[2]; } fb;
  fb.q[0] = p[0];   // K pairs 0..7   (+8*half)
  fb.q[1] = p[2];   // K pairs 16..23 (+8*half)
  return fb.v;
}

// ---------------------------------------------------------------------------
// TDM: DMA one 128x128 bf16 tile (row-major, contiguous in global) into LDS
// with an 8-element (16B) pad after each 128-element (256B = 64-DWORD) row,
// i.e. LDS row stride 136 bf16. One instruction moves the whole tile.
//   D# group0: count=1 | lds_addr | global_addr | type=2
//   D# group1: data_size=2B, pad_enable, pad_interval=64DW(code 5),
//              pad_amount=4DW(code 3), tensor_dim0/1=128, tile_dim0/1=128,
//              tensor_dim0_stride=128
// ---------------------------------------------------------------------------
__device__ __forceinline__ void tdm_load_tile_bf16(const __bf16* gsrc,
                                                   __bf16* ldst) {
  unsigned lds_addr = (unsigned)(unsigned long long)ldst;  // low 32b = LDS off
  unsigned long long ga = (unsigned long long)gsrc;
  u32x4 g0;
  g0[0] = 1u;                                   // count=1, user descriptor
  g0[1] = lds_addr;                             // bits [63:32]
  g0[2] = (unsigned)(ga & 0xFFFFFFFFu);         // global_addr[31:0]
  g0[3] = (unsigned)((ga >> 32) & 0x1FFFFFFu)   // global_addr[56:32]
          | (2u << 30);                         // type=2 ("image")
  i32x8 g1;
  g1[0] = (int)((1u << 16)      // data_size = 2 bytes
              | (1u << 20)      // pad_enable
              | (5u << 22)      // pad_interval: 64 DWORDs
              | (3u << 25));    // pad_amount:   4 DWORDs (16B)
  g1[1] = (int)(128u << 16);    // tensor_dim0[15:0] @ bits[63:48]
  g1[2] = (int)(128u << 16);    // tensor_dim1[15:0] @ bits[95:80]
  g1[3] = (int)(128u << 16);    // tile_dim0 @ bits[127:112]
  g1[4] = 128;                  // tile_dim1 @ bits[143:128]
  g1[5] = 128;                  // tensor_dim0_stride[31:0] @ bits[191:160]
  g1[6] = 0;
  g1[7] = 0;
  i32x4 z4 = (i32x4){0, 0, 0, 0};               // groups 2/3 unused (2D)
  i32x8 z8 = (i32x8){0, 0, 0, 0, 0, 0, 0, 0};
  __builtin_amdgcn_tensor_load_to_lds(g0, g1, z4, z4, z8, 0);
}

// ---------------------------------------------------------------------------
// Projection GEMM:  out[b,h,i,d](bf16) = (X[32768,1024] @ W[1024,1024] + bias)
// Tile: 128x64 per block, 8 waves, each wave 32x32 (2x2 WMMA tiles), BK=32.
// ---------------------------------------------------------------------------
#define GBM 128
#define GBN 64
#define GBK 32
#define LDA 40   // padded bf16 row stride (avoids bank conflicts)

__global__ __launch_bounds__(256) void proj_gemm_kernel(
    const float* __restrict__ A, const float* __restrict__ W,
    const float* __restrict__ bias, __bf16* __restrict__ outb) {
  __shared__ __bf16 sA[GBM * LDA];
  __shared__ __bf16 sBt[GBN * LDA];   // W tile stored N-major: sBt[n][k]

  const int m0 = blockIdx.x * GBM;
  const int n0 = blockIdx.y * GBN;
  const int tid = threadIdx.x;
  const int w = tid >> 5, lane = tid & 31;
  const int wm = (w >> 1) * 32, wn = (w & 1) * 32;

  v8f acc[2][2];
  acc[0][0] = (v8f){}; acc[0][1] = (v8f){};
  acc[1][0] = (v8f){}; acc[1][1] = (v8f){};

  for (int kk = 0; kk < N_DM; kk += GBK) {
    // A tile (fp32 -> bf16): 128x32, float4 loads, packed uint2 LDS stores
#pragma unroll
    for (int it = 0; it < 4; ++it) {
      int idx = tid + it * 256;
      int r = idx >> 3, c4 = idx & 7;
      v4f v = *(const v4f*)(A + (size_t)(m0 + r) * N_DM + kk + c4 * 4);
      union { __bf16 h[4]; v2u u; } pk;
      pk.h[0] = (__bf16)v[0]; pk.h[1] = (__bf16)v[1];
      pk.h[2] = (__bf16)v[2]; pk.h[3] = (__bf16)v[3];
      *(v2u*)(sA + r * LDA + c4 * 4) = pk.u;
    }
    // W tile transposed into N-major (so B fragments load row-wise)
#pragma unroll
    for (int it = 0; it < 2; ++it) {
      int idx = tid + it * 256;
      int kr = idx >> 4, c4 = idx & 15;
      v4f v = *(const v4f*)(W + (size_t)(kk + kr) * N_DM + n0 + c4 * 4);
      sBt[(c4 * 4 + 0) * LDA + kr] = (__bf16)v[0];
      sBt[(c4 * 4 + 1) * LDA + kr] = (__bf16)v[1];
      sBt[(c4 * 4 + 2) * LDA + kr] = (__bf16)v[2];
      sBt[(c4 * 4 + 3) * LDA + kr] = (__bf16)v[3];
    }
    __syncthreads();

    v16bf a0 = ldfrag(sA, LDA, wm, 0);
    v16bf a1 = ldfrag(sA, LDA, wm + 16, 0);
    v16bf b0 = ldfrag(sBt, LDA, wn, 0);
    v16bf b1 = ldfrag(sBt, LDA, wn + 16, 0);
    acc[0][0] = wmma_bf16(a0, b0, acc[0][0]);
    acc[0][1] = wmma_bf16(a0, b1, acc[0][1]);
    acc[1][0] = wmma_bf16(a1, b0, acc[1][0]);
    acc[1][1] = wmma_bf16(a1, b1, acc[1][1]);
    __syncthreads();
  }

  const int hf = lane >> 4, ncol = lane & 15;
#pragma unroll
  for (int mt = 0; mt < 2; ++mt)
#pragma unroll
    for (int nt = 0; nt < 2; ++nt)
#pragma unroll
      for (int r = 0; r < 8; ++r) {
        int m = m0 + wm + mt * 16 + r + 8 * hf;   // = b*S + i
        int n = n0 + wn + nt * 16 + ncol;         // = h*DK + d
        float val = acc[mt][nt][r] + bias[n];
        int b = m >> 7, i = m & 127, h = n >> 7, d = n & 127;
        outb[(((size_t)(b * N_H + h)) * N_S + i) * N_DK + d] = (__bf16)val;
      }
}

// ---------------------------------------------------------------------------
// Output GEMM: out[m,n](f32) = ctx_bf16[32768,1024] @ Wo + bo
// ---------------------------------------------------------------------------
__global__ __launch_bounds__(256) void out_gemm_kernel(
    const __bf16* __restrict__ A, const float* __restrict__ W,
    const float* __restrict__ bias, float* __restrict__ out) {
  __shared__ __bf16 sA[GBM * LDA];
  __shared__ __bf16 sBt[GBN * LDA];

  const int m0 = blockIdx.x * GBM;
  const int n0 = blockIdx.y * GBN;
  const int tid = threadIdx.x;
  const int w = tid >> 5, lane = tid & 31;
  const int wm = (w >> 1) * 32, wn = (w & 1) * 32;

  v8f acc[2][2];
  acc[0][0] = (v8f){}; acc[0][1] = (v8f){};
  acc[1][0] = (v8f){}; acc[1][1] = (v8f){};

  for (int kk = 0; kk < N_DM; kk += GBK) {
    // A tile already bf16: straight b128 copies
#pragma unroll
    for (int it = 0; it < 2; ++it) {
      int idx = tid + it * 256;
      int r = idx >> 2, c8 = idx & 3;
      *(v4u*)(sA + r * LDA + c8 * 8) =
          *(const v4u*)(A + (size_t)(m0 + r) * N_DM + kk + c8 * 8);
    }
#pragma unroll
    for (int it = 0; it < 2; ++it) {
      int idx = tid + it * 256;
      int kr = idx >> 4, c4 = idx & 15;
      v4f v = *(const v4f*)(W + (size_t)(kk + kr) * N_DM + n0 + c4 * 4);
      sBt[(c4 * 4 + 0) * LDA + kr] = (__bf16)v[0];
      sBt[(c4 * 4 + 1) * LDA + kr] = (__bf16)v[1];
      sBt[(c4 * 4 + 2) * LDA + kr] = (__bf16)v[2];
      sBt[(c4 * 4 + 3) * LDA + kr] = (__bf16)v[3];
    }
    __syncthreads();

    v16bf a0 = ldfrag(sA, LDA, wm, 0);
    v16bf a1 = ldfrag(sA, LDA, wm + 16, 0);
    v16bf b0 = ldfrag(sBt, LDA, wn, 0);
    v16bf b1 = ldfrag(sBt, LDA, wn + 16, 0);
    acc[0][0] = wmma_bf16(a0, b0, acc[0][0]);
    acc[0][1] = wmma_bf16(a0, b1, acc[0][1]);
    acc[1][0] = wmma_bf16(a1, b0, acc[1][0]);
    acc[1][1] = wmma_bf16(a1, b1, acc[1][1]);
    __syncthreads();
  }

  const int hf = lane >> 4, ncol = lane & 15;
#pragma unroll
  for (int mt = 0; mt < 2; ++mt)
#pragma unroll
    for (int nt = 0; nt < 2; ++nt)
#pragma unroll
      for (int r = 0; r < 8; ++r) {
        int m = m0 + wm + mt * 16 + r + 8 * hf;
        int n = n0 + wn + nt * 16 + ncol;
        out[(size_t)m * N_DM + n] = acc[mt][nt][r] + bias[n];
      }
}

// ---------------------------------------------------------------------------
// Fused attention: one block per (b,h); everything stays in LDS (~236 KB).
// ---------------------------------------------------------------------------
#define SD  136   // padded bf16 row stride (matches TDM pad descriptor)
#define SFS 132   // padded f32 row stride (scores / pos_full)
#define ATTN_LDS_BYTES (5 * N_S * SD * 2 + N_S * SFS * 4)

__device__ __forceinline__ void load_tile_f32_bf16(const float* __restrict__ g,
                                                   __bf16* sm) {
  int tid = threadIdx.x;
#pragma unroll
  for (int it = 0; it < 16; ++it) {
    int idx = tid + it * 256;          // 4096 x float4
    int r = idx >> 5, c = idx & 31;
    v4f v = *(const v4f*)(g + r * N_DK + c * 4);
    union { __bf16 h[4]; v2u u; } pk;
    pk.h[0] = (__bf16)v[0]; pk.h[1] = (__bf16)v[1];
    pk.h[2] = (__bf16)v[2]; pk.h[3] = (__bf16)v[3];
    *(v2u*)(sm + r * SD + c * 4) = pk.u;
  }
}

// Transposed copies (so v / rel_v act as N-major B operands: sm[d][j])
__device__ __forceinline__ void copy_tile_bf16_T(const __bf16* __restrict__ g,
                                                 __bf16* sm) {
  int tid = threadIdx.x;
#pragma unroll
  for (int it = 0; it < 32; ++it) {
    int idx = tid + it * 256;          // 8192 x b32
    int j = idx >> 6, c = idx & 63;
    unsigned int u = *(const unsigned int*)(g + j * N_DK + c * 2);
    const __bf16* ph = (const __bf16*)&u;
    sm[(c * 2 + 0) * SD + j] = ph[0];
    sm[(c * 2 + 1) * SD + j] = ph[1];
  }
}

__device__ __forceinline__ void load_tile_f32_bf16_T(const float* __restrict__ g,
                                                     __bf16* sm) {
  int tid = threadIdx.x;
#pragma unroll
  for (int it = 0; it < 64; ++it) {
    int idx = tid + it * 256;          // 16384 elements
    int j = idx >> 7, d = idx & 127;
    sm[d * SD + j] = (__bf16)g[j * N_DK + d];
  }
}

__global__ __launch_bounds__(256) void attn_kernel(
    const __bf16* __restrict__ qg, const __bf16* __restrict__ kg,
    const __bf16* __restrict__ vg, const float* __restrict__ relq,
    const float* __restrict__ relk, const float* __restrict__ relv,
    const int* __restrict__ qpos, const int* __restrict__ kvpos,
    __bf16* __restrict__ ctxg) {
  extern __shared__ char smem_raw[];
  __bf16* sK   = (__bf16*)smem_raw;        // k rows; later v^T
  __bf16* sQ   = sK + N_S * SD;            // q rows; later probs
  __bf16* sR   = sQ + N_S * SD;            // rel_q -> rel_k -> rel_v^T
  __bf16* sP2C = sR + N_S * SD;            // p2c table [p][j]
  __bf16* sC2P = sP2C + N_S * SD;          // c2p table [i][p]
  float*  sSc  = (float*)(sC2P + N_S * SD);// c2c scores; later pos_full

  const int bh = blockIdx.x;
  const int b = bh >> 3, h = bh & 7;
  const int tid = threadIdx.x;
  const int w = tid >> 5, lane = tid & 31;
  const int hf = lane >> 4, ncol = lane & 15;
  const int r0 = w * 16;                   // this wave's 16 output rows

  const __bf16* qgb = qg + (size_t)bh * (N_S * N_DK);
  const __bf16* kgb = kg + (size_t)bh * (N_S * N_DK);
  const __bf16* vgb = vg + (size_t)bh * (N_S * N_DK);
  const int* qposb  = qpos  + (size_t)bh * (N_S * N_S);
  const int* kvposb = kvpos + (size_t)bh * (N_S * N_S);

  // Warm L2/WGP$ for the gather-phase index streams (2 x 64KB per block):
  __builtin_prefetch(qposb + tid * 64, 0, 3);   // 256 thr x 256B = 64KB
  __builtin_prefetch(kvposb + tid * 64, 0, 3);

  // ---- phase 1: TDM k and q tiles into padded LDS; rel_q via VALU convert -
  if (__builtin_amdgcn_readfirstlane(w) == 0) {
    tdm_load_tile_bf16(kgb, sK);   // TENSORcnt -> 1
    tdm_load_tile_bf16(qgb, sQ);   // TENSORcnt -> 2 (completes in order)
  }
  load_tile_f32_bf16(relq + (size_t)h * (N_S * N_DK), sR);
  __builtin_amdgcn_s_wait_tensorcnt(1);    // k tile resident (q may fly on)
  __syncthreads();

  // ---- phase 2: p2c table = rel_q @ k^T -----------------------------------
  {
    v16bf aF[4];
#pragma unroll
    for (int ks = 0; ks < 4; ++ks) aF[ks] = ldfrag(sR, SD, r0, ks * 32);
#pragma unroll
    for (int nt = 0; nt < 8; ++nt) {
      v8f a = (v8f){};
#pragma unroll
      for (int ks = 0; ks < 4; ++ks)
        a = wmma_bf16(aF[ks], ldfrag(sK, SD, nt * 16, ks * 32), a);
#pragma unroll
      for (int r = 0; r < 8; ++r)
        sP2C[(r0 + r + 8 * hf) * SD + nt * 16 + ncol] = (__bf16)a[r];
    }
  }
  __syncthreads();

  // ---- phase 3: rel_k; c2c scores + c2p table (q DMA'd in phase 1) --------
  load_tile_f32_bf16(relk + (size_t)h * (N_S * N_DK), sR);
  __builtin_amdgcn_s_wait_tensorcnt(0);    // q tile resident
  __syncthreads();
  {
    v16bf aQ[4];
#pragma unroll
    for (int ks = 0; ks < 4; ++ks) aQ[ks] = ldfrag(sQ, SD, r0, ks * 32);
#pragma unroll
    for (int nt = 0; nt < 8; ++nt) {       // c2c = q @ k^T
      v8f a = (v8f){};
#pragma unroll
      for (int ks = 0; ks < 4; ++ks)
        a = wmma_bf16(aQ[ks], ldfrag(sK, SD, nt * 16, ks * 32), a);
#pragma unroll
      for (int r = 0; r < 8; ++r)
        sSc[(r0 + r + 8 * hf) * SFS + nt * 16 + ncol] = a[r];
    }
#pragma unroll
    for (int nt = 0; nt < 8; ++nt) {       // c2p = q @ rel_k^T
      v8f a = (v8f){};
#pragma unroll
      for (int ks = 0; ks < 4; ++ks)
        a = wmma_bf16(aQ[ks], ldfrag(sR, SD, nt * 16, ks * 32), a);
#pragma unroll
      for (int r = 0; r < 8; ++r)
        sC2P[(r0 + r + 8 * hf) * SD + nt * 16 + ncol] = (__bf16)a[r];
    }
  }
  __syncthreads();

  // ---- phase 4: gather + mask + softmax (wave handles its 16 rows) --------
  for (int rr = 0; rr < 16; ++rr) {
    int i = r0 + rr;
    int j0 = lane * 4;
    v4i qp = *(const v4i*)(qposb + i * N_S + j0);
    v4i kp = *(const v4i*)(kvposb + i * N_S + j0);
    float s[4];
#pragma unroll
    for (int t = 0; t < 4; ++t) {
      int j = j0 + t;
      float v = sSc[i * SFS + j] + (float)sP2C[qp[t] * SD + j] +
                (float)sC2P[i * SD + kp[t]];
      s[t] = v * SCALE_F;
    }
    float mx = fmaxf(fmaxf(s[0], s[1]), fmaxf(s[2], s[3]));
#pragma unroll
    for (int off = 16; off > 0; off >>= 1)
      mx = fmaxf(mx, __shfl_xor(mx, off, 32));
    float e[4], sum = 0.f;
#pragma unroll
    for (int t = 0; t < 4; ++t) {
      bool keep = (i == 0) || (qp[t] == 0);
      e[t] = keep ? __expf(s[t] - mx) : 0.f;
      sum += e[t];
    }
#pragma unroll
    for (int off = 16; off > 0; off >>= 1) sum += __shfl_xor(sum, off, 32);
    float inv  = (sum > 0.f) ? (1.0f / sum) : 0.f;
    float unif = (sum > 0.f) ? 0.f : (1.0f / (float)N_S);  // all-masked row
#pragma unroll
    for (int t = 0; t < 4; ++t)
      sQ[i * SD + j0 + t] = (__bf16)(e[t] * inv + unif);   // probs reuse sQ
  }

  // ---- phase 5: v^T and rel_v^T into LDS ----------------------------------
  copy_tile_bf16_T(vgb, sK);
  load_tile_f32_bf16_T(relv + (size_t)h * (N_S * N_DK), sR);
  __syncthreads();

  // ---- phase 6: pos_full = probs @ rel_v; ctx = probs @ v + gather --------
  {
    v16bf aP[4];
#pragma unroll
    for (int ks = 0; ks < 4; ++ks) aP[ks] = ldfrag(sQ, SD, r0, ks * 32);
#pragma unroll
    for (int nt = 0; nt < 8; ++nt) {       // pos_full -> sSc (reuse)
      v8f a = (v8f){};
#pragma unroll
      for (int ks = 0; ks < 4; ++ks)
        a = wmma_bf16(aP[ks], ldfrag(sR, SD, nt * 16, ks * 32), a);
#pragma unroll
      for (int r = 0; r < 8; ++r)
        sSc[(r0 + r + 8 * hf) * SFS + nt * 16 + ncol] = a[r];
    }
#pragma unroll
    for (int nt = 0; nt < 8; ++nt) {       // ctx tile, add gathered pos ctx
      v8f a = (v8f){};
#pragma unroll
      for (int ks = 0; ks < 4; ++ks)
        a = wmma_bf16(aP[ks], ldfrag(sK, SD, nt * 16, ks * 32), a);
#pragma unroll
      for (int r = 0; r < 8; ++r) {
        int i = r0 + r + 8 * hf;
        int d = nt * 16 + ncol;
        int kv = kvposb[i * N_S + d];
        float val = a[r] + sSc[i * SFS + kv];
        // ctx layout [b, i, h, d] -> row-major [32768, 1024] for out-proj
        ctxg[(((size_t)(b * N_S + i)) * N_H + h) * N_DK + d] = (__bf16)val;
      }
    }
  }
}

// ---------------------------------------------------------------------------
// Launch
// ---------------------------------------------------------------------------
extern "C" void kernel_launch(void* const* d_in, const int* in_sizes, int n_in,
                              void* d_out, int out_size, void* d_ws,
                              size_t ws_size, hipStream_t stream) {
  const float* query = (const float*)d_in[0];
  const float* key   = (const float*)d_in[1];
  const float* value = (const float*)d_in[2];
  const int*   qpos  = (const int*)d_in[3];
  const int*   kvpos = (const int*)d_in[4];
  const float* relq  = (const float*)d_in[5];
  const float* relk  = (const float*)d_in[6];
  const float* relv  = (const float*)d_in[7];
  const float* Wq = (const float*)d_in[8];
  const float* bq = (const float*)d_in[9];
  const float* Wk = (const float*)d_in[10];
  const float* bk = (const float*)d_in[11];
  const float* Wv = (const float*)d_in[12];
  const float* bv = (const float*)d_in[13];
  const float* Wo = (const float*)d_in[14];
  const float* bo = (const float*)d_in[15];
  float* out = (float*)d_out;

  const size_t NQKV = (size_t)N_B * N_S * N_DM;  // 33,554,432 elements
  __bf16* qb = (__bf16*)d_ws;       // [B,H,S,DK] bf16
  __bf16* kb = qb + NQKV;
  __bf16* vb = kb + NQKV;
  __bf16* cb = vb + NQKV;           // ctx [B,S,H*DK] bf16

  dim3 gg(N_B * N_S / GBM, N_DM / GBN);  // (256, 16)
  dim3 blk(256);

  proj_gemm_kernel<<<gg, blk, 0, stream>>>(query, Wq, bq, qb);
  proj_gemm_kernel<<<gg, blk, 0, stream>>>(key,   Wk, bk, kb);
  proj_gemm_kernel<<<gg, blk, 0, stream>>>(value, Wv, bv, vb);

  attn_kernel<<<dim3(N_B * N_H), blk, ATTN_LDS_BYTES, stream>>>(
      qb, kb, vb, relq, relk, relv, qpos, kvpos, cb);

  out_gemm_kernel<<<gg, blk, 0, stream>>>(cb, Wo, bo, out);
}
